// RecurrentNeuralNetwork_5437428596986
// MI455X (gfx1250) — matrix operand order repaired and checked by reading, not compiled, not run on hardware
//
#include <hip/hip_runtime.h>

#define B_DIM 64
#define T_DIM 512
#define I_DIM 512
#define H_DIM 2048
#define O_DIM 512
#define NWG_REC 64
#define WAVES_PER_BLOCK 8

typedef __attribute__((ext_vector_type(16))) _Float16 v16h;
typedef __attribute__((ext_vector_type(8)))  float    v8f;
typedef __attribute__((ext_vector_type(4)))  unsigned int u32x4;
typedef __attribute__((ext_vector_type(8)))  int          i32x8;
typedef __attribute__((ext_vector_type(4)))  int          i32x4;

union Frag {
  uint4    q[2];
  _Float16 h[16];
  v16h     v;
};

// ---- A-fragment (16x32 f16, MxK) from row-major [16 x ld] fp32, cvt on the fly ----
// ISA 7.12.2: lanes 0-15 hold row M=lane, K chunks {0-7,16-23}; lanes 16-31 {8-15,24-31}
static __device__ inline v16h load_a_frag_f32(const float* __restrict__ base, int ld,
                                              int k, int lane) {
  const int r    = lane & 15;
  const int koff = (lane >> 4) << 3;            // 0 or 8
  const float* p0 = base + r * ld + k + koff;   // K .. K+7
  const float* p1 = p0 + 16;                    // K+16 .. K+23
  Frag f;
#pragma unroll
  for (int i = 0; i < 8; ++i) f.h[i]     = (_Float16)p0[i];
#pragma unroll
  for (int i = 0; i < 8; ++i) f.h[8 + i] = (_Float16)p1[i];
  return f.v;
}

// ---- A-fragment (16x32 f16) from row-major [16 x ld] f16 ----
static __device__ inline v16h load_a_frag_f16(const _Float16* __restrict__ base, int ld,
                                              int k, int lane) {
  const int r    = lane & 15;
  const int koff = (lane >> 4) << 3;            // 0 or 8
  const _Float16* p0 = base + r * ld + k + koff;
  Frag f;
  f.q[0] = *(const uint4*)(p0);                 // K..K+7   (or 8..15)
  f.q[1] = *(const uint4*)(p0 + 16);            // K+16..23 (or 24..31)
  return f.v;
}

// ---- B-fragment (32x16 f16, KxN) from row-major W[N][K] f16 (column-major B) ----
// lanes 0-15: column N=lane, K {0-7,16-23}; lanes 16-31: K {8-15,24-31}
// Works for global or LDS pointers (addrspace inferred after inlining -> ds_load_b128)
static __device__ inline v16h load_b_frag_f16(const _Float16* w, int ld, int k, int lane) {
  const int n    = lane & 15;
  const int koff = (lane >> 4) << 3;            // 0 or 8
  const _Float16* p = w + n * ld + k + koff;
  Frag f;
  f.q[0] = *(const uint4*)(p);
  f.q[1] = *(const uint4*)(p + 16);
  return f.v;
}

// ---- TDM descriptor builders (ISA cdna5 §8, D# groups) ----
static __device__ inline u32x4 tdm_group0(unsigned lds_byte_off,
                                          unsigned long long gaddr) {
  u32x4 g;
  g.x = 1u;                                             // count=1, user descriptor
  g.y = lds_byte_off;                                   // lds_addr [63:32]
  g.z = (unsigned)(gaddr & 0xFFFFFFFFull);              // global_addr [95:64]
  g.w = (unsigned)((gaddr >> 32) & 0x1FFFFFFull)        // global_addr [120:96]
        | (2u << 30);                                   // type=2 ("image")
  return g;
}

union TDG1 { int w[8]; i32x8 v; };

// 2D tile of 16-bit elements: rows x cols, row stride = stride0 elements
static __device__ inline i32x8 tdm_group1_2d_f16(unsigned cols, unsigned rows,
                                                 unsigned stride0) {
  TDG1 g = {};
  g.w[0] = (int)(1u << 16);                             // data_size=1 (2 bytes)
  g.w[1] = (int)((cols & 0xFFFFu) << 16);               // tensor_dim0[15:0]
  g.w[2] = (int)(((cols >> 16) & 0xFFFFu) | ((rows & 0xFFFFu) << 16)); // dim0 hi, dim1 lo
  g.w[3] = (int)(((rows >> 16) & 0xFFFFu) | ((cols & 0xFFFFu) << 16)); // dim1 hi, tile_dim0
  g.w[4] = (int)(rows & 0xFFFFu);                       // tile_dim1 (tile_dim2=0)
  g.w[5] = (int)stride0;                                // tensor_dim0_stride lo32
  g.w[6] = 0;                                           // stride0 hi, stride1 lo
  g.w[7] = 0;
  return g.v;
}

// ---- monotonic grid-wide barrier (persistent kernel, one signal per WG) ----
static __device__ inline void grid_barrier(unsigned* counter, unsigned target) {
  __threadfence();                              // publish our stores device-wide
  __syncthreads();
  if (threadIdx.x == 0) {
    atomicAdd(counter, 1u);
    while (__hip_atomic_load(counter, __ATOMIC_ACQUIRE, __HIP_MEMORY_SCOPE_AGENT) < target) {
      __builtin_amdgcn_s_sleep(2);
    }
  }
  __syncthreads();
}

// ---------------------------------------------------------------------------
// xih = x @ W_ih^T + b_ih -> hs slot of d_out ([B*T, H]); wave = 16x64 slab.
// Register double-buffered K loop so loads overlap the WMMAs.
// ---------------------------------------------------------------------------
__global__ __launch_bounds__(256) void xih_gemm_kernel(
    const float* __restrict__ x, const _Float16* __restrict__ Wih,
    const float* __restrict__ bih, float* __restrict__ hs) {
  const int lane = threadIdx.x & 31;
  const int wave = blockIdx.x * WAVES_PER_BLOCK + (threadIdx.x >> 5);
  const int mt = wave >> 5;                     // 0..2047 (M tiles of B*T=32768)
  const int ng = wave & 31;                     // N group of 4 tiles
  const int m0 = mt << 4;
  const int n0 = ng << 6;
  const float* abase = x + (size_t)m0 * I_DIM;

  v8f c[4] = {};
  v16h a = load_a_frag_f32(abase, I_DIM, 0, lane);
  v16h b[4], b2[4];
#pragma unroll
  for (int j = 0; j < 4; ++j)
    b[j] = load_b_frag_f16(Wih + (size_t)(n0 + j * 16) * I_DIM, I_DIM, 0, lane);

#pragma unroll 2
  for (int k = 0; k < I_DIM; k += 32) {
    v16h a2;
    if (k + 32 < I_DIM) {                       // prefetch next K block
      a2 = load_a_frag_f32(abase, I_DIM, k + 32, lane);
#pragma unroll
      for (int j = 0; j < 4; ++j)
        b2[j] = load_b_frag_f16(Wih + (size_t)(n0 + j * 16) * I_DIM, I_DIM, k + 32, lane);
    }
#pragma unroll
    for (int j = 0; j < 4; ++j)
      c[j] = __builtin_amdgcn_wmma_f32_16x16x32_f16(false, a, false, b[j],
                                                    (short)0, c[j], false, false);
    a = a2;
#pragma unroll
    for (int j = 0; j < 4; ++j) b[j] = b2[j];
  }

  const int n  = lane & 15;
  const int mb = (lane >> 4) << 3;
#pragma unroll
  for (int j = 0; j < 4; ++j) {
    const int   nn   = n0 + j * 16 + n;
    const float bias = bih[nn];
#pragma unroll
    for (int v = 0; v < 8; ++v)
      hs[(size_t)(m0 + mb + v) * H_DIM + nn] = c[j][v] + bias;  // pre-activation
  }
}

// ---------------------------------------------------------------------------
// Persistent recurrence: 64 WGs x 8 waves. WG wg owns h-tiles {mt 0..3} x
// {n0 = 32wg, 32wg+16} and y-tiles {2wg, 2wg+1}. W_hh / W_ho slabs (32 rows
// each, 128 KB + 128 KB f16) staged into LDS ONCE by the Tensor Data Mover;
// inner loop = ds_load + wmma. h state double-buffered in f16 in ws
// (L2-resident); one grid barrier per timestep.
// ---------------------------------------------------------------------------
__global__ __launch_bounds__(256) void rnn_recurrence_kernel(
    const _Float16* __restrict__ Whh, const float* __restrict__ bhh,
    const _Float16* __restrict__ Who, const float* __restrict__ bho,
    float* __restrict__ hs, float* __restrict__ ys,
    _Float16* __restrict__ hbuf, unsigned* __restrict__ counter) {
  extern __shared__ _Float16 smem[];            // 262144 B dynamic LDS
  _Float16* WhhS = smem;                        // [32][H_DIM] at LDS offset 0
  _Float16* WhoS = smem + 32 * H_DIM;           // [32][H_DIM] at LDS offset 131072

  const int lane = threadIdx.x & 31;
  const int w    = threadIdx.x >> 5;            // wave in WG: 0..7
  const int wg   = blockIdx.x;                  // 0..63

  // ---- TDM: stage loop-invariant weight slabs into LDS (wave 0 issues) ----
  if (w == 0) {
    const unsigned long long ga_h =
        (unsigned long long)(uintptr_t)(Whh + (size_t)(wg * 32) * H_DIM);
    const unsigned long long ga_o =
        (unsigned long long)(uintptr_t)(Who + (size_t)(((2 * wg) & 31) * 16) * H_DIM);
    const i32x8 g1 = tdm_group1_2d_f16(H_DIM, 32, H_DIM);   // 32 rows x 2048 f16
    const i32x4 gz = {0, 0, 0, 0};                          // <=2D: groups 2/3 unused
    const i32x8 gz8 = {0, 0, 0, 0, 0, 0, 0, 0};
    __builtin_amdgcn_tensor_load_to_lds(tdm_group0(0u, ga_h), g1, gz, gz, gz8, 0);
    __builtin_amdgcn_tensor_load_to_lds(tdm_group0(131072u, ga_o), g1, gz, gz, gz8, 0);
    __builtin_amdgcn_s_wait_tensorcnt(0);
  }
  __syncthreads();

  // ---- tile assignment ----
  const int mt  = w & 3;
  const int ntl = w >> 2;                       // 0..1 (local n tile)
  const int m0  = mt << 4;
  const int n0g = wg * 32 + ntl * 16;           // global n
  const _Float16* wldsH = WhhS + ntl * 16 * H_DIM;

  const bool do_y = (w == 0) || (w == 4);
  const int  ytl  = w >> 2;                     // 0 or 1
  const int  ym0  = ((2 * wg) >> 5) << 4;
  const int  yn0g = ((2 * wg + ytl) & 31) << 4;
  const _Float16* wldsY = WhoS + ytl * 16 * H_DIM;

  const int n  = lane & 15;
  const int mb = (lane >> 4) << 3;
  const float bias_h = bhh[n0g + n];
  const float bias_y = bho[yn0g + n];
  const size_t BH = (size_t)B_DIM * H_DIM;

  for (int t = 0; t < T_DIM; ++t) {
    const _Float16* hprev = hbuf + (size_t)((t & 1) ^ 1) * BH;
    _Float16*       hcur  = hbuf + (size_t)(t & 1) * BH;

    // ---- hidden-state GEMM tile: c = h_{t-1}[m0:,:] @ Whh[n0:,:]^T ----
    v8f c = {};
    if (t > 0) {
      const _Float16* abase = hprev + (size_t)m0 * H_DIM;
      v16h a = load_a_frag_f16(abase, H_DIM, 0, lane);
      v16h b = load_b_frag_f16(wldsH, H_DIM, 0, lane);
#pragma unroll 2
      for (int k = 0; k < H_DIM - 32; k += 32) {
        v16h a2 = load_a_frag_f16(abase, H_DIM, k + 32, lane);
        v16h b2 = load_b_frag_f16(wldsH, H_DIM, k + 32, lane);
        c = __builtin_amdgcn_wmma_f32_16x16x32_f16(false, a, false, b,
                                                   (short)0, c, false, false);
        a = a2; b = b2;
      }
      c = __builtin_amdgcn_wmma_f32_16x16x32_f16(false, a, false, b,
                                                 (short)0, c, false, false);
    }

    // ---- epilogue: RMW xih (already in hs), + bias, ReLU; fp32 out + f16 state ----
#pragma unroll
    for (int v = 0; v < 8; ++v) {
      float* p = hs + ((size_t)(m0 + mb + v) * T_DIM + t) * H_DIM + n0g + n;
      float val = *p + c[v] + bias_h;
      val = val > 0.f ? val : 0.f;
      *p = val;
      hcur[(size_t)(m0 + mb + v) * H_DIM + n0g + n] = (_Float16)val;
    }

    // ---- publish h_t to all WGPs ----
    grid_barrier(counter, (unsigned)(NWG_REC * (t + 1)));

    // ---- output head (overlaps next step's GEMM on the other 6 waves) ----
    if (do_y) {
      v8f cy = {};
      const _Float16* abase = hcur + (size_t)ym0 * H_DIM;
      v16h a = load_a_frag_f16(abase, H_DIM, 0, lane);
      v16h b = load_b_frag_f16(wldsY, H_DIM, 0, lane);
#pragma unroll 2
      for (int k = 0; k < H_DIM - 32; k += 32) {
        v16h a2 = load_a_frag_f16(abase, H_DIM, k + 32, lane);
        v16h b2 = load_b_frag_f16(wldsY, H_DIM, k + 32, lane);
        cy = __builtin_amdgcn_wmma_f32_16x16x32_f16(false, a, false, b,
                                                    (short)0, cy, false, false);
        a = a2; b = b2;
      }
      cy = __builtin_amdgcn_wmma_f32_16x16x32_f16(false, a, false, b,
                                                  (short)0, cy, false, false);
#pragma unroll
      for (int v = 0; v < 8; ++v) {
        float val = cy[v] + bias_y;
        ys[((size_t)(ym0 + mb + v) * T_DIM + t) * O_DIM + yn0g + n] =
            val > 0.f ? val : 0.f;
      }
    }
  }
}

// ---------------------------------------------------------------------------
__global__ void convert_f32_to_f16_kernel(const float* __restrict__ in,
                                          _Float16* __restrict__ out, int nelem) {
  int i = blockIdx.x * blockDim.x + threadIdx.x;
  if (i < nelem) out[i] = (_Float16)in[i];
}

__global__ void init_counter_kernel(unsigned* c) {
  if (threadIdx.x == 0 && blockIdx.x == 0) *c = 0u;
}

extern "C" void kernel_launch(void* const* d_in, const int* in_sizes, int n_in,
                              void* d_out, int out_size, void* d_ws, size_t ws_size,
                              hipStream_t stream) {
  const float* x   = (const float*)d_in[0];
  const float* Wih = (const float*)d_in[1];
  const float* bih = (const float*)d_in[2];
  const float* Whh = (const float*)d_in[3];
  const float* bhh = (const float*)d_in[4];
  const float* Who = (const float*)d_in[5];
  const float* bho = (const float*)d_in[6];

  float* hs = (float*)d_out;                                  // [B,T,H]
  float* ys = hs + (size_t)B_DIM * T_DIM * H_DIM;             // [B,T,O]

  // workspace layout
  char* ws = (char*)d_ws;
  unsigned*  counter = (unsigned*)ws;                                   // 256 B
  _Float16*  Wih16   = (_Float16*)(ws + 256);                           // 2 MB
  _Float16*  Whh16   = Wih16 + (size_t)H_DIM * I_DIM;                   // 8 MB
  _Float16*  Who16   = Whh16 + (size_t)H_DIM * H_DIM;                   // 2 MB
  _Float16*  hbuf    = Who16 + (size_t)O_DIM * H_DIM;                   // 0.5 MB (x2 buf)

  const int nWih = H_DIM * I_DIM;
  const int nWhh = H_DIM * H_DIM;
  const int nWho = O_DIM * H_DIM;
  convert_f32_to_f16_kernel<<<(nWih + 255) / 256, 256, 0, stream>>>(Wih, Wih16, nWih);
  convert_f32_to_f16_kernel<<<(nWhh + 255) / 256, 256, 0, stream>>>(Whh, Whh16, nWhh);
  convert_f32_to_f16_kernel<<<(nWho + 255) / 256, 256, 0, stream>>>(Who, Who16, nWho);
  init_counter_kernel<<<1, 64, 0, stream>>>(counter);

  // Phase A: xih into the hs slot of d_out. 2048 M-tiles x 32 N-groups = 65536 waves
  xih_gemm_kernel<<<65536 / WAVES_PER_BLOCK, 256, 0, stream>>>(x, Wih16, bih, hs);

  // Phase B: persistent sequential recurrence + output head (256 KB dynamic LDS)
  rnn_recurrence_kernel<<<NWG_REC, 256, 2 * 32 * H_DIM * sizeof(_Float16), stream>>>(
      Whh16, bhh, Who16, bho, hs, ys, hbuf, counter);
}
